// FourierHead2DLR_35639638622762
// MI455X (gfx1250) — compile-verified, matrix operand order checked by reading, not run on hardware
//
#include <hip/hip_runtime.h>
#include <hip/hip_bf16.h>

typedef __attribute__((ext_vector_type(8)))  _Float16 v8h;
typedef __attribute__((ext_vector_type(16))) _Float16 v16h;
typedef __attribute__((ext_vector_type(8)))  float    v8f;
typedef __attribute__((ext_vector_type(4)))  float    v4f;

constexpr int BROWS     = 16384;   // batch
constexpr int KDIM      = 512;     // reduction dim
constexpr int NCOLS     = 1056;    // 4 * RANK * L
constexpr int NT_TOTAL  = 72;      // padded to 8 waves * 9 tiles
constexpr int NCOLS_PAD = NT_TOTAL * 16;   // 1152
constexpr int NT_W      = 9;       // n-tiles per wave
constexpr int LJ        = 33;      // L = M+1

// ---------------- W (1056x512 f32) -> padded (1152x512 f16) ----------------
__global__ __launch_bounds__(256) void cvt_w_f16(const float* __restrict__ W,
                                                 _Float16* __restrict__ Wh) {
  int idx = blockIdx.x * 256 + threadIdx.x;      // 0 .. 1152*512-1
  float v = (idx < NCOLS * KDIM) ? W[idx] : 0.0f;
  Wh[idx] = (_Float16)v;
}

// ---------------- fused GEMM (WMMA f16) + Fourier-head epilogue ------------
__global__ __launch_bounds__(256) void fourier_head_fused(
    const float*    __restrict__ x,     // (B, 512) f32
    const float*    __restrict__ tgt,   // (B, 2)   f32
    const _Float16* __restrict__ Wh,    // (1152, 512) f16 (padded)
    const float*    __restrict__ bias,  // (1056,) f32
    float*          __restrict__ out)   // (B,) f32
{
  __shared__ _Float16 lds_x[16 * KDIM];            // 16 KB
  __shared__ float    lds_z[16 * NCOLS_PAD];       // 72 KB
  __shared__ float    red_pq[128];
  __shared__ float    red_cn[128];

  const int tid      = threadIdx.x;
  const int wave     = tid >> 5;
  const int ln       = tid & 31;
  const int row_base = blockIdx.x * 16;

  // ---- stage 16 rows of x into LDS as f16 (flat 8192-float copy) ----
  const float* xb = x + (size_t)row_base * KDIM;
  #pragma unroll
  for (int c = 0; c < 8; ++c) {
    int g = c * 256 + tid;                 // float4 index, 2048 total
    v4f v = *(const v4f*)(xb + (size_t)g * 4);
    _Float16* d = lds_x + g * 4;
    d[0] = (_Float16)v[0]; d[1] = (_Float16)v[1];
    d[2] = (_Float16)v[2]; d[3] = (_Float16)v[3];
  }
  __syncthreads();

  // ---- WMMA GEMM: 16 rows x (9 tiles * 16 cols) per wave, K = 512 ----
  const int r  = ln & 15;      // A-row / B-col within tile
  const int hi = ln >> 4;      // K-half selector per ISA 16-bit layout

  v8f acc[NT_W];
  #pragma unroll
  for (int i = 0; i < NT_W; ++i)
    #pragma unroll
    for (int v = 0; v < 8; ++v) acc[i][v] = 0.0f;

  const _Float16* aRow = lds_x + r * KDIM + hi * 8;
  const _Float16* bRow = Wh + ((size_t)(wave * NT_W * 16 + r)) * KDIM + hi * 8;

  #pragma unroll 2
  for (int k0 = 0; k0 < KDIM; k0 += 32) {
    // A fragment: halves [k0+hi*8 .. +7] and [k0+16+hi*8 .. +7]
    v8h alo = *(const v8h*)(aRow + k0);
    v8h ahi = *(const v8h*)(aRow + k0 + 16);
    v16h A = __builtin_shufflevector(alo, ahi,
        0,1,2,3,4,5,6,7,8,9,10,11,12,13,14,15);
    #pragma unroll
    for (int i = 0; i < NT_W; ++i) {
      const _Float16* bp = bRow + (size_t)i * 16 * KDIM + k0;
      v8h blo = *(const v8h*)(bp);
      v8h bhi = *(const v8h*)(bp + 16);
      v16h Bm = __builtin_shufflevector(blo, bhi,
          0,1,2,3,4,5,6,7,8,9,10,11,12,13,14,15);
      acc[i] = __builtin_amdgcn_wmma_f32_16x16x32_f16(
          false, A, false, Bm, (short)0, acc[i], false, false);
    }
  }

  // ---- scatter z-tile (+bias) to LDS: lane holds col=r, rows v+8*hi ----
  #pragma unroll
  for (int i = 0; i < NT_W; ++i) {
    int col = (wave * NT_W + i) * 16 + r;
    float bv = (col < NCOLS) ? bias[col] : 0.0f;
    float* zp = lds_z + (hi * 8) * NCOLS_PAD + col;
    #pragma unroll
    for (int v = 0; v < 8; ++v)
      zp[v * NCOLS_PAD] = acc[i][v] + bv;
  }
  __syncthreads();

  // ---- epilogue: thread = (row, rank); P_r = |Sum_j a_j e^{i pi t_y j}|^2 ----
  if (tid < 128) {
    int row = tid >> 3, rr = tid & 7;
    int b   = row_base + row;
    float tx = tgt[2 * b + 0];
    float ty = tgt[2 * b + 1];
    float sy, cy; sincospif(ty, &sy, &cy);
    float sx, cx; sincospif(tx, &sx, &cx);

    const float* zr = lds_z + row * NCOLS_PAD + rr * LJ;
    const float* z0 = zr;              // comp 0 (Re a)
    const float* z1 = zr + 264;        // comp 1 (Im a)
    const float* z2 = zr + 528;        // comp 2 (Re b)
    const float* z3 = zr + 792;        // comp 3 (Im b)

    float car = 1.f, cai = 0.f, sar = 0.f, sai = 0.f, na = 0.f;
    float cbr = 1.f, cbi = 0.f, sbr = 0.f, sbi = 0.f, nb = 0.f;
    #pragma unroll 1
    for (int j = 0; j < LJ; ++j) {
      float a0 = z0[j], a1 = z1[j];
      sar = fmaf(a0, car, fmaf(-a1, cai, sar));
      sai = fmaf(a0, cai, fmaf( a1, car, sai));
      na  = fmaf(a0, a0, fmaf(a1, a1, na));
      float t1 = car * cy - cai * sy;
      cai = car * sy + cai * cy; car = t1;

      float b0 = z2[j], b1 = z3[j];
      sbr = fmaf(b0, cbr, fmaf(-b1, cbi, sbr));
      sbi = fmaf(b0, cbi, fmaf( b1, cbr, sbi));
      nb  = fmaf(b0, b0, fmaf(b1, b1, nb));
      float t2 = cbr * cx - cbi * sx;
      cbi = cbr * sx + cbi * cx; cbr = t2;
    }
    float P = sar * sar + sai * sai;
    float Q = sbr * sbr + sbi * sbi;
    red_pq[tid] = P * Q;
    red_cn[tid] = na * nb;
  }
  __syncthreads();

  // ---- reduce over ranks, normalize, clamp ----
  if (tid < 16) {
    float spq = 0.f, scn = 0.f;
    #pragma unroll
    for (int i = 0; i < 8; ++i) {
      spq += red_pq[tid * 8 + i];
      scn += red_cn[tid * 8 + i];
    }
    float pdf = spq / (4.0f * fmaxf(scn, 0.01f));   // sign(centre) == +1
    out[row_base + tid] = fmaxf(pdf, 1e-8f);
  }
}

extern "C" void kernel_launch(void* const* d_in, const int* in_sizes, int n_in,
                              void* d_out, int out_size, void* d_ws, size_t ws_size,
                              hipStream_t stream) {
  const float* x    = (const float*)d_in[0];   // (16384, 512)
  const float* tgt  = (const float*)d_in[1];   // (16384, 2)
  const float* W    = (const float*)d_in[2];   // (1056, 512)
  const float* bias = (const float*)d_in[3];   // (1056,)
  float* out = (float*)d_out;

  _Float16* Wh = (_Float16*)d_ws;              // 1152*512*2 = 1.18 MB scratch

  const int wtotal = NCOLS_PAD * KDIM;         // 589824
  cvt_w_f16<<<wtotal / 256, 256, 0, stream>>>(W, Wh);
  fourier_head_fused<<<BROWS / 16, 256, 0, stream>>>(x, tgt, Wh, bias, out);
}
